// WindowAttention_11948599018020
// MI455X (gfx1250) — matrix operand order repaired and checked by reading, not compiled, not run on hardware
//
#include <hip/hip_runtime.h>
#include <stdint.h>

// ---------------------------------------------------------------------------
// Types / WMMA helpers
// ---------------------------------------------------------------------------
typedef __attribute__((ext_vector_type(16))) _Float16 v16h;
typedef __attribute__((ext_vector_type(8)))  float    v8f;

struct Frag { union { v16h h; uint4 q[2]; }; };

__device__ __forceinline__ v8f wmma16(const Frag& a, const Frag& b, v8f c) {
    // D = A(16x32 f16) x B(32x16 f16) + C(16x16 f32)
    return __builtin_amdgcn_wmma_f32_16x16x32_f16(
        /*neg_a=*/false, a.h, /*neg_b=*/false, b.h,
        /*c_mod=*/(short)0, c, /*reuse_a=*/false, /*reuse_b=*/false);
}

// CDNA5 async DMA: per-lane 16B global -> LDS, tracked by ASYNCcnt.
// The "VDST" operand is the wave-relative LDS byte offset, which for a
// generic pointer to __shared__ is the low 32 address bits (flat->LDS rule).
__device__ __forceinline__ void async_load_b128(const void* gaddr,
                                                uint32_t lds_off) {
    asm volatile("global_load_async_to_lds_b128 %0, %1, off"
                 :: "v"(lds_off), "v"(gaddr)
                 : "memory");
}
__device__ __forceinline__ void wait_async0() {
    asm volatile("s_wait_asynccnt 0x0" ::: "memory");
}
__device__ __forceinline__ uint32_t lds_off_of(const void* p) {
    return (uint32_t)(uint64_t)p;
}

// LDS tile row pitch in f16 elements: 48 -> 96B rows, every 16B-access aligned
#define LPITCH 48

// A-operand fragment: lane holds row (lane&15), K vals {k0..k0+7, k0+16..k0+23}
__device__ __forceinline__ Frag ldsFragA(const _Float16* tile, int row, int lane) {
    Frag f;
    const int k0 = (lane >> 4) << 3;
    const uint4* p = (const uint4*)(tile + row * LPITCH + k0);
    f.q[0] = p[0];      // K = k0 .. k0+7
    f.q[1] = p[2];      // K = k0+16 .. k0+23
    return f;
}
// B-operand fragment: lane holds col (lane&15), K vals {k0..k0+7, k0+16..k0+23}
__device__ __forceinline__ Frag ldsFragB(const _Float16* tile, int row, int lane) {
    Frag f;
    const int kb = (lane >> 4) << 3;
    const uint4* p = (const uint4*)(tile + row * LPITCH + kb);
    f.q[0] = p[0];
    f.q[1] = p[2];
    return f;
}
__device__ __forceinline__ Frag glbFragA(const _Float16* rowbase, int lane) {
    Frag f;
    const int k0 = (lane >> 4) << 3;
    f.q[0] = *(const uint4*)(rowbase + k0);
    f.q[1] = *(const uint4*)(rowbase + k0 + 16);
    return f;
}
__device__ __forceinline__ Frag glbFragB(const _Float16* rowbase, int lane) {
    Frag f;
    const int kb = (lane >> 4) << 3;
    const uint4* p = (const uint4*)(rowbase + kb);
    f.q[0] = p[0];
    f.q[1] = p[2];
    return f;
}

// ---------------------------------------------------------------------------
// Prep kernels
// ---------------------------------------------------------------------------
__global__ void f32_to_f16_kernel(const float* __restrict__ src,
                                  _Float16* __restrict__ dst, int n) {
    int i = blockIdx.x * blockDim.x + threadIdx.x;
    if (i < n) dst[i] = (_Float16)src[i];
}

// biasT[h*4096 + i*64 + j] = table[idx[i,j]][h]; scaleT likewise with [16+h]
__global__ void build_tab_kernel(const float* __restrict__ table,
                                 const int* __restrict__ idx,
                                 float* __restrict__ biasT,
                                 float* __restrict__ scaleT) {
    int t  = blockIdx.x * blockDim.x + threadIdx.x;   // 16*4096
    int h  = t >> 12;
    int ij = t & 4095;
    int id = idx[ij];
    biasT[t]  = table[id * 32 + h];
    scaleT[t] = table[id * 32 + 16 + h];
}

// ---------------------------------------------------------------------------
// Tiled WMMA GEMM:  C(MxN) = A(MxK) * B(NxK)^T   (both operands K-major)
// block = 256 thr = 8 waves; macro tile 128x128x32; wave tile 64x32
// MODE 0: A is f32 (x), epilogue scatters QKV (q scaled, v transposed)
// MODE 1: A is f16 (attn out), epilogue writes f32 + bias
// f16 tiles are staged with global_load_async_to_lds_b128 (ASYNCcnt path);
// the f32 A tile (MODE 0) is reg-staged because it needs an f32->f16 convert.
// ---------------------------------------------------------------------------
union H4 { _Float16 h[4]; uint2 u; };

__device__ __forceinline__ void fetch_a32(float4* r, const float* A, int bm,
                                          int ks, int K, int tid) {
#pragma unroll
    for (int i = 0; i < 4; ++i) {
        int v = tid + 256 * i;                 // 0..1023
        int row = v >> 3, col = (v & 7) << 2;  // 128 x 32 f32
        r[i] = *(const float4*)(A + (size_t)(bm + row) * K + ks + col);
    }
}
__device__ __forceinline__ void stash_a32(_Float16* S, const float4* r, int tid) {
#pragma unroll
    for (int i = 0; i < 4; ++i) {
        int v = tid + 256 * i;
        int row = v >> 3, col = (v & 7) << 2;
        H4 x;
        x.h[0] = (_Float16)r[i].x; x.h[1] = (_Float16)r[i].y;
        x.h[2] = (_Float16)r[i].z; x.h[3] = (_Float16)r[i].w;
        *(uint2*)(S + row * LPITCH + col) = x.u;
    }
}
// issue async copies of one 128x32 f16 tile (K-major rows) into LDS
__device__ __forceinline__ void issue_tile_h(_Float16* S, const _Float16* G,
                                             int b0, int ks, int K, int tid) {
#pragma unroll
    for (int i = 0; i < 2; ++i) {
        int v = tid + 256 * i;                 // 0..511
        int row = v >> 2, col = (v & 3) << 3;  // 128 x 32 f16, 16B chunks
        async_load_b128(G + (size_t)(b0 + row) * K + ks + col,
                        lds_off_of(S + row * LPITCH + col));
    }
}

template <int MODE>
__global__ __launch_bounds__(256) void gemm_kernel(
    const void* __restrict__ Ag_, const _Float16* __restrict__ Bg,
    const float* __restrict__ bias, void* o0, void* o1, void* o2,
    int K, int N) {
    __shared__ _Float16 As[2][128 * LPITCH];
    __shared__ _Float16 Bs[2][128 * LPITCH];

    const int tid = threadIdx.x;
    const int bm = blockIdx.x * 128;
    const int bn = blockIdx.y * 128;
    const int lane = tid & 31, wv = tid >> 5;
    const int wm = (wv >> 2) * 64, wn = (wv & 3) * 32;
    const int r0 = lane & 15;

    const float*    Af = (const float*)Ag_;
    const _Float16* Ah = (const _Float16*)Ag_;

    v8f acc[4][2];
#pragma unroll
    for (int mt = 0; mt < 4; ++mt) {
        v8f z = {};
        acc[mt][0] = z; acc[mt][1] = z;
    }

    // prologue: tile 0 -> buf 0
    if constexpr (MODE == 0) {
        float4 r[4];
        fetch_a32(r, Af, bm, 0, K, tid);
        stash_a32(As[0], r, tid);
    } else {
        issue_tile_h(As[0], Ah, bm, 0, K, tid);
    }
    issue_tile_h(Bs[0], Bg, bn, 0, K, tid);
    wait_async0();
    __syncthreads();

    const int NS = K >> 5;
    for (int s = 0; s < NS; ++s) {
        const int cur = s & 1;
        const bool nxt = (s + 1) < NS;
        float4 ra[4];
        if (nxt) {  // overlap next tile's DMA / loads with this tile's WMMAs
            const int ks = (s + 1) << 5;
            if constexpr (MODE == 0) fetch_a32(ra, Af, bm, ks, K, tid);
            else                     issue_tile_h(As[cur ^ 1], Ah, bm, ks, K, tid);
            issue_tile_h(Bs[cur ^ 1], Bg, bn, ks, K, tid);
        }
        Frag af[4], bf[2];
#pragma unroll
        for (int mt = 0; mt < 4; ++mt)
            af[mt] = ldsFragA(&As[cur][0], wm + mt * 16 + r0, lane);
#pragma unroll
        for (int nt = 0; nt < 2; ++nt)
            bf[nt] = ldsFragB(&Bs[cur][0], wn + nt * 16 + r0, lane);
#pragma unroll
        for (int mt = 0; mt < 4; ++mt)
#pragma unroll
            for (int nt = 0; nt < 2; ++nt)
                acc[mt][nt] = wmma16(af[mt], bf[nt], acc[mt][nt]);
        if (nxt) {
            if constexpr (MODE == 0) stash_a32(As[cur ^ 1], ra, tid);
            wait_async0();
            __syncthreads();
        }
    }

    // epilogue -- C layout: VGPR i -> M = i + 8*(lane>=16), N = lane&15
    const int halfo = (lane >> 4) << 3;
    if constexpr (MODE == 0) {
        const float qscale = 0.17677669529663687f;  // 32^-0.5
        _Float16* Qo = (_Float16*)o0;
        _Float16* Ko = (_Float16*)o1;
        _Float16* Vo = (_Float16*)o2;
#pragma unroll
        for (int mt = 0; mt < 4; ++mt)
#pragma unroll
            for (int nt = 0; nt < 2; ++nt) {
                const int n = bn + wn + nt * 16 + r0;
                const float bv = bias[n];
                const int mb = bm + wm + mt * 16 + halfo;
#pragma unroll
                for (int i = 0; i < 8; ++i) {
                    const int m = mb + i;
                    const float v = acc[mt][nt][i] + bv;
                    const int b = m >> 6, r = m & 63;
                    if (n < 512) {         // Q: (b,h,n,d), pre-scaled
                        const int h = n >> 5, d = n & 31;
                        Qo[(((size_t)b * 16 + h) * 64 + r) * 32 + d] =
                            (_Float16)(v * qscale);
                    } else if (n < 1024) { // K: (b,h,n,d)
                        const int h = (n - 512) >> 5, d = n & 31;
                        Ko[(((size_t)b * 16 + h) * 64 + r) * 32 + d] = (_Float16)v;
                    } else {               // V transposed: (b,h,d,n)
                        const int h = (n - 1024) >> 5, d = n & 31;
                        Vo[(((size_t)b * 16 + h) * 32 + d) * 64 + r] = (_Float16)v;
                    }
                }
            }
    } else {
        float* Out = (float*)o0;
#pragma unroll
        for (int mt = 0; mt < 4; ++mt)
#pragma unroll
            for (int nt = 0; nt < 2; ++nt) {
                const int n = bn + wn + nt * 16 + r0;
                const float bv = bias[n];
                const int mb = bm + wm + mt * 16 + halfo;
#pragma unroll
                for (int i = 0; i < 8; ++i)
                    Out[(size_t)(mb + i) * N + n] = acc[mt][nt][i] + bv;
            }
    }
}

// ---------------------------------------------------------------------------
// Attention: one block (4 waves) per (b,h); wave handles a 16-row strip.
// S = Q K^T (4 WMMAs, K=hd=32) -> +bias -> softmax -> *scale -> LDS ->
// O = P V^T-form (4 WMMAs, K=64) -> f16 out (b, n, h*32+d)
// ---------------------------------------------------------------------------
#define PP 72  // P strip row pitch (f16): 144B rows, 16B aligned accesses

__global__ __launch_bounds__(128) void attn_kernel(
    const _Float16* __restrict__ Q, const _Float16* __restrict__ Kb,
    const _Float16* __restrict__ Vt, const float* __restrict__ biasT,
    const float* __restrict__ scaleT, _Float16* __restrict__ O) {
    __shared__ _Float16 P[4][16 * PP];

    const int tid = threadIdx.x, lane = tid & 31, w = tid >> 5;
    const int b = blockIdx.x >> 4, h = blockIdx.x & 15;
    const int r0 = lane & 15;
    const int halfo = (lane >> 4) << 3;

    const size_t headQK = ((size_t)(b * 16 + h)) * 64 * 32;
    const size_t headV  = ((size_t)(b * 16 + h)) * 32 * 64;

    // S = Q * K^T (per-strip: 16x64 in 4 c-fragments)
    Frag qf = glbFragA(Q + headQK + (size_t)(w * 16 + r0) * 32, lane);
    v8f s[4];
#pragma unroll
    for (int nt = 0; nt < 4; ++nt) {
        Frag kf = glbFragB(Kb + headQK + (size_t)(nt * 16 + r0) * 32, lane);
        v8f z = {};
        s[nt] = wmma16(qf, kf, z);
    }

    // relative bias + softmax + relative scale
    const float* bT = biasT + h * 4096;
    const float* sT = scaleT + h * 4096;
    const int mrow0 = w * 16 + halfo;
#pragma unroll
    for (int nt = 0; nt < 4; ++nt)
#pragma unroll
        for (int i = 0; i < 8; ++i)
            s[nt][i] += bT[(mrow0 + i) * 64 + nt * 16 + r0];

#pragma unroll
    for (int i = 0; i < 8; ++i) {
        float mx = fmaxf(fmaxf(s[0][i], s[1][i]), fmaxf(s[2][i], s[3][i]));
        mx = fmaxf(mx, __shfl_xor(mx, 1, 32));
        mx = fmaxf(mx, __shfl_xor(mx, 2, 32));
        mx = fmaxf(mx, __shfl_xor(mx, 4, 32));
        mx = fmaxf(mx, __shfl_xor(mx, 8, 32));
        float sum = 0.f;
#pragma unroll
        for (int nt = 0; nt < 4; ++nt) {
            float e = __expf(s[nt][i] - mx);
            s[nt][i] = e;
            sum += e;
        }
        sum += __shfl_xor(sum, 1, 32);
        sum += __shfl_xor(sum, 2, 32);
        sum += __shfl_xor(sum, 4, 32);
        sum += __shfl_xor(sum, 8, 32);
        const float inv = 1.0f / sum;
#pragma unroll
        for (int nt = 0; nt < 4; ++nt) {
            float p = s[nt][i] * inv * sT[(mrow0 + i) * 64 + nt * 16 + r0];
            // C-layout -> row-major strip in LDS (A-fragment source)
            P[w][(halfo + i) * PP + nt * 16 + r0] = (_Float16)p;
        }
    }
    __syncthreads();

    // O = P * V^T-form  (A = P strip 16x64, B = Vt rows j-contiguous)
    v8f o0 = {}, o1 = {};
#pragma unroll
    for (int jc = 0; jc < 2; ++jc) {
        Frag pf;
        {
            const int k0 = (lane >> 4) << 3;
            const uint4* p = (const uint4*)(&P[w][r0 * PP + jc * 32 + k0]);
            pf.q[0] = p[0];
            pf.q[1] = p[2];
        }
        Frag vf0 = glbFragB(Vt + headV + (size_t)(r0) * 64 + jc * 32, lane);
        Frag vf1 = glbFragB(Vt + headV + (size_t)(16 + r0) * 64 + jc * 32, lane);
        o0 = wmma16(pf, vf0, o0);
        o1 = wmma16(pf, vf1, o1);
    }
#pragma unroll
    for (int i = 0; i < 8; ++i) {
        const int m = w * 16 + halfo + i;
        const size_t base = ((size_t)b * 64 + m) * 512 + h * 32;
        O[base + r0]      = (_Float16)o0[i];
        O[base + 16 + r0] = (_Float16)o1[i];
    }
}

// ---------------------------------------------------------------------------
// Launch
// ---------------------------------------------------------------------------
extern "C" void kernel_launch(void* const* d_in, const int* in_sizes, int n_in,
                              void* d_out, int out_size, void* d_ws,
                              size_t ws_size, hipStream_t stream) {
    (void)in_sizes; (void)n_in; (void)out_size; (void)ws_size;

    const float* x      = (const float*)d_in[0];  // (65536, 512)
    const float* qkv_w  = (const float*)d_in[1];  // (1536, 512)
    const float* qkv_b  = (const float*)d_in[2];  // (1536,)
    const float* proj_w = (const float*)d_in[3];  // (512, 512)
    const float* proj_b = (const float*)d_in[4];  // (512,)
    const float* rtab   = (const float*)d_in[5];  // (225, 32)
    const int*   ridx   = (const int*)d_in[6];    // (64, 64)

    char* w = (char*)d_ws;
    auto carve = [&](size_t bytes) {
        char* p = w;
        w += (bytes + 255) & ~(size_t)255;
        return (void*)p;
    };
    _Float16* wqkv_h  = (_Float16*)carve((size_t)1536 * 512 * 2);
    _Float16* wproj_h = (_Float16*)carve((size_t)512 * 512 * 2);
    float*    biasT   = (float*)carve((size_t)16 * 4096 * 4);
    float*    scaleT  = (float*)carve((size_t)16 * 4096 * 4);
    const size_t qkvElems = (size_t)1024 * 16 * 64 * 32;  // 33.5M
    _Float16* Qb = (_Float16*)carve(qkvElems * 2);
    _Float16* Kb = (_Float16*)carve(qkvElems * 2);
    _Float16* Vt = (_Float16*)carve(qkvElems * 2);
    _Float16* Ob = (_Float16*)carve((size_t)1024 * 64 * 512 * 2);

    // prep
    f32_to_f16_kernel<<<(786432 + 255) / 256, 256, 0, stream>>>(qkv_w, wqkv_h, 786432);
    f32_to_f16_kernel<<<(262144 + 255) / 256, 256, 0, stream>>>(proj_w, wproj_h, 262144);
    build_tab_kernel<<<65536 / 256, 256, 0, stream>>>(rtab, ridx, biasT, scaleT);

    // QKV projection: (65536 x 512) * (1536 x 512)^T, scatter Q/K/Vt (f16)
    gemm_kernel<0><<<dim3(512, 12), 256, 0, stream>>>(
        x, wqkv_h, qkv_b, Qb, Kb, Vt, 512, 1536);

    // windowed attention, one block per (b, h)
    attn_kernel<<<1024 * 16, 128, 0, stream>>>(Qb, Kb, Vt, biasT, scaleT, Ob);

    // output projection: (65536 x 512) * (512 x 512)^T + bias -> f32
    gemm_kernel<1><<<dim3(512, 4), 256, 0, stream>>>(
        Ob, wproj_h, proj_b, d_out, nullptr, nullptr, 512, 512);
}
